// LearnedQueryAttention_4844723110071
// MI455X (gfx1250) — matrix operand order repaired and checked by reading, not compiled, not run on hardware
//
#include <hip/hip_runtime.h>
#include <hip/hip_bf16.h>

// Problem constants (from reference)
#define BATCH 4
#define SEQ   4096
#define NQ    256
#define DM    1024
#define NH    16
#define HD    64

typedef __attribute__((ext_vector_type(16))) __bf16 v16bf;
typedef __attribute__((ext_vector_type(8)))  float  v8f;
typedef unsigned short u16t;

// ---------- bf16 <-> f32 helpers (round-to-nearest-even) ----------
__device__ __forceinline__ u16t f2bf(float f) {
  unsigned u = __float_as_uint(f);
  unsigned r = 0x7FFFu + ((u >> 16) & 1u);
  return (u16t)((u + r) >> 16);
}
__device__ __forceinline__ float bf2f(u16t h) {
  return __uint_as_float(((unsigned)h) << 16);
}

union FragU { uint4 u4[2]; v16bf v; };

// ---------- WMMA fragment loaders (ISA 7.12.2 layouts, wave32) ----------
// A (16x32 bf16, M x K): lane m = lane&15, hi = lane>>4.
//   elements 0..7  = K: k0 + hi*8 .. +7
//   elements 8..15 = K: k0 + 16 + hi*8 .. +7
__device__ __forceinline__ v16bf load_a_frag(const u16t* base, int ld, int m0, int k0, int lane) {
  const int m  = m0 + (lane & 15);
  const int hi = lane >> 4;
  const u16t* p = base + (size_t)m * ld + (k0 + hi * 8);
  FragU f;
  f.u4[0] = *reinterpret_cast<const uint4*>(p);
  f.u4[1] = *reinterpret_cast<const uint4*>(p + 16);
  return f.v;
}

// B (32x16 bf16, K x N), source stored row-major as Brow[n][k]:
//   lane n = lane&15, hi = lane>>4; elements 0..7 = K: k0 + hi*8 .. +7, elements 8..15 = K: k0 + 16 + hi*8 .. +7
__device__ __forceinline__ v16bf load_b_frag(const u16t* base, int ld, int n0, int k0, int lane) {
  const int n  = n0 + (lane & 15);
  const int hi = lane >> 4;
  const u16t* p = base + (size_t)n * ld + (k0 + hi * 8);
  FragU f;
  f.u4[0] = *reinterpret_cast<const uint4*>(p);
  f.u4[1] = *reinterpret_cast<const uint4*>(p + 16);
  return f.v;
}

__device__ __forceinline__ v8f wmma_bf16(v16bf a, v16bf b, v8f c) {
  return __builtin_amdgcn_wmma_f32_16x16x32_bf16(false, a, false, b, (short)0, c, false, false);
}

// C/D (16x16 f32): lane n = lane&15, element r -> M = m0 + r + (lane>>4)*8
__device__ __forceinline__ void store_tile_f32(float* C, int ldc, int m0, int n0, int lane, v8f acc, float scale) {
  const int n  = n0 + (lane & 15);
  const int mb = m0 + (lane >> 4) * 8;
  #pragma unroll
  for (int r = 0; r < 8; ++r) C[(size_t)(mb + r) * ldc + n] = acc[r] * scale;
}
__device__ __forceinline__ void store_tile_bf16(u16t* C, int ldc, int m0, int n0, int lane, v8f acc) {
  const int n  = n0 + (lane & 15);
  const int mb = m0 + (lane >> 4) * 8;
  #pragma unroll
  for (int r = 0; r < 8; ++r) C[(size_t)(mb + r) * ldc + n] = f2bf(acc[r]);
}
// Transposed store: Ct[n][m], contiguous in m -> each lane's 8 accumulator values are
// 8 consecutive M at fixed N == one 16-byte store.
__device__ __forceinline__ void store_tile_bf16_T(u16t* Ct, int ldt, int m0, int n0, int lane, v8f acc) {
  const int n  = n0 + (lane & 15);
  const int mb = m0 + (lane >> 4) * 8;
  union { u16t s[8]; uint4 u; } pk;
  #pragma unroll
  for (int r = 0; r < 8; ++r) pk.s[r] = f2bf(acc[r]);
  *reinterpret_cast<uint4*>(Ct + (size_t)n * ldt + mb) = pk.u;
}

// ---------- elementwise f32 -> bf16 ----------
__global__ void __launch_bounds__(256)
f32_to_bf16_kernel(const float* __restrict__ X, u16t* __restrict__ Y, int n) {
  int i = blockIdx.x * 256 + threadIdx.x;
  if (i < n) Y[i] = f2bf(X[i]);
}

// ---------- layernorm (row of DM=1024, 256 threads) -> bf16 ----------
__global__ void __launch_bounds__(256)
layernorm_to_bf16(const float* __restrict__ X, const float* __restrict__ g,
                  const float* __restrict__ beta, u16t* __restrict__ Y) {
  __shared__ float sh[256];
  const size_t row = blockIdx.x;
  const float* x = X + row * DM;
  float v[4];
  float lsum = 0.f;
  #pragma unroll
  for (int i = 0; i < 4; ++i) { v[i] = x[threadIdx.x + i * 256]; lsum += v[i]; }
  sh[threadIdx.x] = lsum; __syncthreads();
  for (int o = 128; o > 0; o >>= 1) {
    if ((int)threadIdx.x < o) sh[threadIdx.x] += sh[threadIdx.x + o];
    __syncthreads();
  }
  const float mu = sh[0] * (1.f / DM); __syncthreads();
  float lss = 0.f;
  #pragma unroll
  for (int i = 0; i < 4; ++i) { float d = v[i] - mu; lss += d * d; }
  sh[threadIdx.x] = lss; __syncthreads();
  for (int o = 128; o > 0; o >>= 1) {
    if ((int)threadIdx.x < o) sh[threadIdx.x] += sh[threadIdx.x + o];
    __syncthreads();
  }
  const float rstd = rsqrtf(sh[0] * (1.f / DM) + 1e-5f);
  u16t* y = Y + row * DM;
  #pragma unroll
  for (int i = 0; i < 4; ++i) {
    int c = threadIdx.x + i * 256;
    y[c] = f2bf((v[i] - mu) * rstd * g[c] + beta[c]);
  }
}

// ---------- generic GEMM: C[M,N] = A[M,K] * W[N,K]^T  (bf16 in, f32 acc) ----------
// 256 thr = 8 waves, block tile 128M x 64N, wave tile 32x32 (4 WMMA / K-step)
// OUT_MODE: 0 = f32 row-major, 1 = bf16 row-major, 2 = bf16 transposed per batch
//           (Ct[b][n][m'] with m' = m - b*SEQ, row stride SEQ) -- used for v_t.
template <int OUT_MODE>
__global__ void __launch_bounds__(256)
gemm_xwT(const u16t* __restrict__ A, const u16t* __restrict__ W,
         void* __restrict__ C, int M, int N, int K) {
  const int lane = threadIdx.x & 31;
  const int wave = threadIdx.x >> 5;
  const int m0 = blockIdx.y * 128 + (wave >> 1) * 32;
  const int n0 = blockIdx.x * 64  + (wave & 1) * 32;
  v8f c00 = {}, c01 = {}, c10 = {}, c11 = {};
  const u16t* pfA = A + (size_t)(m0 + (lane & 15)) * K;
  const u16t* pfW = W + (size_t)(n0 + (lane & 15)) * K;
  for (int k0 = 0; k0 < K; k0 += 32) {
    if (k0 + 32 < K) {  // uniform branch: prefetch next K-tile (global_prefetch_b8)
      __builtin_prefetch(pfA + k0 + 32, 0, 3);
      __builtin_prefetch(pfW + k0 + 32, 0, 3);
    }
    v16bf a0 = load_a_frag(A, K, m0,      k0, lane);
    v16bf a1 = load_a_frag(A, K, m0 + 16, k0, lane);
    v16bf b0 = load_b_frag(W, K, n0,      k0, lane);
    v16bf b1 = load_b_frag(W, K, n0 + 16, k0, lane);
    c00 = wmma_bf16(a0, b0, c00);
    c01 = wmma_bf16(a0, b1, c01);
    c10 = wmma_bf16(a1, b0, c10);
    c11 = wmma_bf16(a1, b1, c11);
  }
  if (OUT_MODE == 0) {
    float* Cf = (float*)C;
    store_tile_f32(Cf, N, m0,      n0,      lane, c00, 1.f);
    store_tile_f32(Cf, N, m0,      n0 + 16, lane, c01, 1.f);
    store_tile_f32(Cf, N, m0 + 16, n0,      lane, c10, 1.f);
    store_tile_f32(Cf, N, m0 + 16, n0 + 16, lane, c11, 1.f);
  } else if (OUT_MODE == 1) {
    u16t* Cb = (u16t*)C;
    store_tile_bf16(Cb, N, m0,      n0,      lane, c00);
    store_tile_bf16(Cb, N, m0,      n0 + 16, lane, c01);
    store_tile_bf16(Cb, N, m0 + 16, n0,      lane, c10);
    store_tile_bf16(Cb, N, m0 + 16, n0 + 16, lane, c11);
  } else {
    // per-batch transpose: tile never crosses a batch boundary (SEQ % 128 == 0)
    const int b  = m0 / SEQ;
    const int ms = m0 - b * SEQ;
    u16t* Ct = (u16t*)C + (size_t)b * DM * SEQ;
    store_tile_bf16_T(Ct, SEQ, ms,      n0,      lane, c00);
    store_tile_bf16_T(Ct, SEQ, ms,      n0 + 16, lane, c01);
    store_tile_bf16_T(Ct, SEQ, ms + 16, n0,      lane, c10);
    store_tile_bf16_T(Ct, SEQ, ms + 16, n0 + 16, lane, c11);
  }
}

// ---------- scores[b,h,q,s] = (q_h · k_h^T) / sqrt(hd) ----------
// grid: x = S/128, y = Q/64, z = B*H ; block 256 (waves: 2 along Q x 4 along S)
__global__ void __launch_bounds__(256)
scores_kernel(const u16t* __restrict__ qbf, const u16t* __restrict__ kbf,
              float* __restrict__ scores) {
  const int lane = threadIdx.x & 31;
  const int wave = threadIdx.x >> 5;
  const int bh = blockIdx.z;
  const int b = bh >> 4, h = bh & 15;
  const int q0 = blockIdx.y * 64  + (wave >> 2) * 32;
  const int s0 = blockIdx.x * 128 + (wave & 3) * 32;
  const u16t* qb = qbf + (size_t)b * NQ  * DM + h * HD;   // row stride DM, contiguous hd
  const u16t* kb = kbf + (size_t)b * SEQ * DM + h * HD;
  v8f c00 = {}, c01 = {}, c10 = {}, c11 = {};
  #pragma unroll
  for (int k0 = 0; k0 < HD; k0 += 32) {
    v16bf a0 = load_a_frag(qb, DM, q0,      k0, lane);
    v16bf a1 = load_a_frag(qb, DM, q0 + 16, k0, lane);
    v16bf b0 = load_b_frag(kb, DM, s0,      k0, lane);
    v16bf b1 = load_b_frag(kb, DM, s0 + 16, k0, lane);
    c00 = wmma_bf16(a0, b0, c00);
    c01 = wmma_bf16(a0, b1, c01);
    c10 = wmma_bf16(a1, b0, c10);
    c11 = wmma_bf16(a1, b1, c11);
  }
  float* out = scores + (size_t)bh * NQ * SEQ;
  const float sc = 0.125f; // 1/sqrt(64)
  store_tile_f32(out, SEQ, q0,      s0,      lane, c00, sc);
  store_tile_f32(out, SEQ, q0,      s0 + 16, lane, c01, sc);
  store_tile_f32(out, SEQ, q0 + 16, s0,      lane, c10, sc);
  store_tile_f32(out, SEQ, q0 + 16, s0 + 16, lane, c11, sc);
}

// ---------- masked softmax over S per (b,h,q) row; writes bf16 attn ----------
__global__ void __launch_bounds__(256)
softmax_mask_kernel(const float* __restrict__ scores, const unsigned char* __restrict__ amask,
                    const unsigned char* __restrict__ kpm, u16t* __restrict__ attn) {
  __shared__ float sh[256];
  const size_t row = blockIdx.x;                 // (b*H + h)*Q + q
  const int b = (int)(row / (NH * NQ));
  const float* s = scores + row * SEQ;
  const unsigned char* am = amask + row * SEQ;
  const unsigned char* km = kpm + (size_t)b * SEQ;
  float vals[16];
  float vmax = -3.4e38f;
  #pragma unroll
  for (int i = 0; i < 16; ++i) {
    const int c = (int)threadIdx.x + i * 256;
    const bool m = (am[c] != 0) || (km[c] != 0);
    const float x = m ? -__builtin_inff() : s[c];
    vals[i] = x;
    vmax = fmaxf(vmax, x);
  }
  sh[threadIdx.x] = vmax; __syncthreads();
  for (int o = 128; o > 0; o >>= 1) {
    if ((int)threadIdx.x < o) sh[threadIdx.x] = fmaxf(sh[threadIdx.x], sh[threadIdx.x + o]);
    __syncthreads();
  }
  vmax = sh[0]; __syncthreads();
  u16t* arow = attn + row * SEQ;
  if (vmax < -3.0e38f) {                         // row fully masked -> zeros
    #pragma unroll
    for (int i = 0; i < 16; ++i) arow[(int)threadIdx.x + i * 256] = 0;
    return;
  }
  float lsum = 0.f;
  #pragma unroll
  for (int i = 0; i < 16; ++i) {
    const float e = __expf(vals[i] - vmax);      // exp(-inf)=0 handles masked cols
    vals[i] = e;
    lsum += e;
  }
  sh[threadIdx.x] = lsum; __syncthreads();
  for (int o = 128; o > 0; o >>= 1) {
    if ((int)threadIdx.x < o) sh[threadIdx.x] += sh[threadIdx.x + o];
    __syncthreads();
  }
  const float inv = 1.f / sh[0];
  #pragma unroll
  for (int i = 0; i < 16; ++i) arow[(int)threadIdx.x + i * 256] = f2bf(vals[i] * inv);
}

// ---------- ctx[b,q,h,d] = sum_s attn[b,h,q,s] * v_t[b][h*HD+d][s] ----------
// v is stored transposed per batch: v_t[b][n][s], n = h*HD + d, row stride SEQ.
// B-operand columns (fixed d, varying s) are now CONTIGUOUS -> b128 loads.
// grid: y = Q/128, z = B*H ; waves: 4 along Q x 2 along d
__global__ void __launch_bounds__(256)
ctx_kernel(const u16t* __restrict__ attn, const u16t* __restrict__ vt,
           float* __restrict__ ctx) {
  const int lane = threadIdx.x & 31;
  const int wave = threadIdx.x >> 5;
  const int bh = blockIdx.z;
  const int b = bh >> 4, h = bh & 15;
  const int q0 = blockIdx.y * 128 + (wave >> 1) * 32;
  const int d0 = (wave & 1) * 32;
  const u16t* ab = attn + (size_t)bh * NQ * SEQ;
  const u16t* vb = vt + ((size_t)b * DM + h * HD) * SEQ;  // row = d within head, ld = SEQ
  v8f c00 = {}, c01 = {}, c10 = {}, c11 = {};
  const u16t* pfA = ab + (size_t)(q0 + (lane & 15)) * SEQ;
  const u16t* pfB = vb + (size_t)(d0 + (lane & 15)) * SEQ;
  for (int k0 = 0; k0 < SEQ; k0 += 32) {
    if (k0 + 32 < SEQ) {  // uniform prefetch of next K-tile
      __builtin_prefetch(pfA + k0 + 32, 0, 3);
      __builtin_prefetch(pfB + k0 + 32, 0, 3);
    }
    v16bf a0 = load_a_frag(ab, SEQ, q0,      k0, lane);
    v16bf a1 = load_a_frag(ab, SEQ, q0 + 16, k0, lane);
    v16bf b0 = load_b_frag(vb, SEQ, d0,      k0, lane);
    v16bf b1 = load_b_frag(vb, SEQ, d0 + 16, k0, lane);
    c00 = wmma_bf16(a0, b0, c00);
    c01 = wmma_bf16(a0, b1, c01);
    c10 = wmma_bf16(a1, b0, c10);
    c11 = wmma_bf16(a1, b1, c11);
  }
  float* out = ctx + (size_t)b * NQ * DM + h * HD;        // (b,q,h,d) row stride DM
  store_tile_f32(out, DM, q0,      d0,      lane, c00, 1.f);
  store_tile_f32(out, DM, q0,      d0 + 16, lane, c01, 1.f);
  store_tile_f32(out, DM, q0 + 16, d0,      lane, c10, 1.f);
  store_tile_f32(out, DM, q0 + 16, d0 + 16, lane, c11, 1.f);
}

// ---------- attn.mean over heads -> (B,Q,S) f32 ----------
__global__ void __launch_bounds__(256)
attn_mean_kernel(const u16t* __restrict__ attn, float* __restrict__ out) {
  const size_t idx = (size_t)blockIdx.x * 256 + threadIdx.x;  // b*(Q*S) + q*S + s
  const size_t QS = (size_t)NQ * SEQ;
  const int b = (int)(idx / QS);
  const size_t rem = idx - (size_t)b * QS;
  const u16t* base = attn + (size_t)b * NH * QS + rem;
  float acc = 0.f;
  #pragma unroll
  for (int h = 0; h < NH; ++h) acc += bf2f(base[(size_t)h * QS]);
  out[idx] = acc * (1.f / NH);
}

extern "C" void kernel_launch(void* const* d_in, const int* in_sizes, int n_in,
                              void* d_out, int out_size, void* d_ws, size_t ws_size,
                              hipStream_t stream) {
  (void)in_sizes; (void)n_in; (void)out_size; (void)ws_size;
  const float* x   = (const float*)d_in[0];
  const float* qry = (const float*)d_in[1];
  const unsigned char* amask = (const unsigned char*)d_in[2];   // jax bool -> 1 byte
  const unsigned char* kpm   = (const unsigned char*)d_in[3];
  const float* Wq = (const float*)d_in[4];
  const float* Wk = (const float*)d_in[5];
  const float* Wv = (const float*)d_in[6];
  const float* Wo = (const float*)d_in[7];
  const float* g1 = (const float*)d_in[8];
  const float* b1 = (const float*)d_in[9];
  const float* g2 = (const float*)d_in[10];
  const float* b2 = (const float*)d_in[11];

  // workspace carve-out (256B aligned chunks), ~498 MiB total
  char* w = (char*)d_ws;
  size_t off = 0;
  auto alloc = [&](size_t bytes) -> char* {
    char* p = w + off;
    off += (bytes + 255) & ~(size_t)255;
    return p;
  };
  u16t*  xn_bf   = (u16t*)alloc((size_t)BATCH * SEQ * DM * 2);
  u16t*  k_bf    = (u16t*)alloc((size_t)BATCH * SEQ * DM * 2);
  u16t*  vt_bf   = (u16t*)alloc((size_t)BATCH * DM * SEQ * 2);  // transposed v
  u16t*  q_bf    = (u16t*)alloc((size_t)BATCH * NQ  * DM * 2);
  u16t*  qu_bf   = (u16t*)alloc((size_t)BATCH * NQ  * DM * 2);
  u16t*  wq_bf   = (u16t*)alloc((size_t)DM * DM * 2);
  u16t*  wk_bf   = (u16t*)alloc((size_t)DM * DM * 2);
  u16t*  wv_bf   = (u16t*)alloc((size_t)DM * DM * 2);
  u16t*  wo_bf   = (u16t*)alloc((size_t)DM * DM * 2);
  float* scoresW = (float*)alloc((size_t)BATCH * NH * NQ * SEQ * 4);
  u16t*  attnW   = (u16t*)alloc((size_t)BATCH * NH * NQ * SEQ * 2);
  float* ctxW    = (float*)alloc((size_t)BATCH * NQ * DM * 4);
  u16t*  ctxn_bf = (u16t*)alloc((size_t)BATCH * NQ * DM * 2);

  // 1) bf16 conversions for queries + weights
  const int NW = DM * DM;
  f32_to_bf16_kernel<<<(BATCH * NQ * DM + 255) / 256, 256, 0, stream>>>(qry, qu_bf, BATCH * NQ * DM);
  f32_to_bf16_kernel<<<(NW + 255) / 256, 256, 0, stream>>>(Wq, wq_bf, NW);
  f32_to_bf16_kernel<<<(NW + 255) / 256, 256, 0, stream>>>(Wk, wk_bf, NW);
  f32_to_bf16_kernel<<<(NW + 255) / 256, 256, 0, stream>>>(Wv, wv_bf, NW);
  f32_to_bf16_kernel<<<(NW + 255) / 256, 256, 0, stream>>>(Wo, wo_bf, NW);

  // 2) LN1(x) -> xn bf16
  layernorm_to_bf16<<<BATCH * SEQ, 256, 0, stream>>>(x, g1, b1, xn_bf);

  // 3) projections (bf16 WMMA, f32 accumulate)
  gemm_xwT<1><<<dim3(DM / 64, (BATCH * NQ) / 128), 256, 0, stream>>>(qu_bf, wq_bf, q_bf, BATCH * NQ, DM, DM);
  gemm_xwT<1><<<dim3(DM / 64, (BATCH * SEQ) / 128), 256, 0, stream>>>(xn_bf, wk_bf, k_bf, BATCH * SEQ, DM, DM);
  gemm_xwT<2><<<dim3(DM / 64, (BATCH * SEQ) / 128), 256, 0, stream>>>(xn_bf, wv_bf, vt_bf, BATCH * SEQ, DM, DM);

  // 4) scores + masked softmax -> attn (bf16)
  scores_kernel<<<dim3(SEQ / 128, NQ / 64, BATCH * NH), 256, 0, stream>>>(q_bf, k_bf, scoresW);
  softmax_mask_kernel<<<BATCH * NH * NQ, 256, 0, stream>>>(scoresW, amask, kpm, attnW);

  // 5) ctx = attn @ v  (bf16 WMMA, contiguous B loads from v_t)
  ctx_kernel<<<dim3(1, NQ / 128, BATCH * NH), 256, 0, stream>>>(attnW, vt_bf, ctxW);

  // 6) attn mean over heads -> second output segment
  float* outF = (float*)d_out;
  attn_mean_kernel<<<(BATCH * NQ * SEQ) / 256, 256, 0, stream>>>(attnW, outF + (size_t)BATCH * NQ * DM);

  // 7) LN2(ctx) -> bf16, final projection -> first output segment (f32)
  layernorm_to_bf16<<<BATCH * NQ, 256, 0, stream>>>(ctxW, g2, b2, ctxn_bf);
  gemm_xwT<0><<<dim3(DM / 64, (BATCH * NQ) / 128), 256, 0, stream>>>(ctxn_bf, wo_bf, outF, BATCH * NQ, DM, DM);
}
